// SuperGlueNetwork_87067577024959
// MI455X (gfx1250) — matrix operand order repaired and checked by reading, not compiled, not run on hardware
//
#include <hip/hip_runtime.h>
#include <hip/hip_bf16.h>
#include <math.h>

// ---------------------------------------------------------------------------
// SuperGlue forward for MI455X (gfx1250, wave32, WMMA).
// Dense matmuls: v_wmma_f32_16x16x32_bf16.  Fragment LDS layouts arranged so
// every fragment load is 2-4x ds_load_b128.  W tiles staged via
// global_load_async_to_lds_b128 (ASYNCcnt).  Attention fused (online softmax).
// ---------------------------------------------------------------------------

#define Bb   4
#define Nn   2048
#define Dd   256
#define Hh   4
#define DIMH 64
#define PP   2049
#define NEGS 0.01f
#define EPSS 1e-5f
#define THRS 0.2f

#ifndef TRY_ASYNC
#define TRY_ASYNC 1
#endif

typedef __bf16 bf16_t;
typedef __attribute__((ext_vector_type(16))) __bf16 v16bf;
typedef __attribute__((ext_vector_type(8)))  __bf16 v8bf;
typedef __attribute__((ext_vector_type(2)))  __bf16 v2bf;
typedef __attribute__((ext_vector_type(8)))  float  v8f;
typedef __attribute__((ext_vector_type(4)))  float  v4f;
typedef __attribute__((ext_vector_type(2)))  float  v2f;

__device__ inline v8f zero8() {
    v8f z = {0.f, 0.f, 0.f, 0.f, 0.f, 0.f, 0.f, 0.f};
    return z;
}

__device__ inline v8f wmma_bf16(v16bf a, v16bf b, v8f c) {
    return __builtin_amdgcn_wmma_f32_16x16x32_bf16(false, a, false, b,
                                                   (short)0, c, false, false);
}

__device__ inline v16bf cat16(v8bf lo, v8bf hi) {
    return __builtin_shufflevector(lo, hi, 0, 1, 2, 3, 4, 5, 6, 7,
                                   8, 9, 10, 11, 12, 13, 14, 15);
}

// A-matrix fragment (16x32 MxK) from row-major bf16 tile.  Per ISA 7.12.2 a
// lane's 16 elements are the two contiguous 8-element k-runs
// [k0+half*8, +8) and [k0+16+half*8, +8)  ->  2x ds_load_b128.
__device__ inline v16bf afrag_bf16(const bf16_t* T, int ld, int r0, int k0, int lane) {
    int half = (lane >> 4) & 1, r = lane & 15;
    const bf16_t* row = T + (size_t)(r0 + r) * ld + k0 + half * 8;
    v8bf lo = *(const v8bf*)(row);
    v8bf hi = *(const v8bf*)(row + 16);
    return cat16(lo, hi);
}

// Same A fragment but from a raw f32 tile (async-staged); convert in regs.
__device__ inline v16bf afrag_f32(const float* T, int ld, int r0, int k0, int lane) {
    int half = (lane >> 4) & 1, r = lane & 15;
    const float* row = T + (size_t)(r0 + r) * ld + k0 + half * 8;
    const v4f* p0 = (const v4f*)row;
    const v4f* p1 = (const v4f*)(row + 16);
    v4f a0 = p0[0], a1 = p0[1], b0 = p1[0], b1 = p1[1];
    v16bf a;
#pragma unroll
    for (int i = 0; i < 4; ++i) {
        a[i]      = (bf16_t)a0[i];
        a[4 + i]  = (bf16_t)a1[i];
        a[8 + i]  = (bf16_t)b0[i];
        a[12 + i] = (bf16_t)b1[i];
    }
    return a;
}

// B-matrix fragment (32x16 KxN) from a TRANSPOSED tile TT[col][k] (k contig).
// A lane's 16 elements are k-runs [k0+half*8, +8) and [k0+16+half*8, +8) of one column.
__device__ inline v16bf bfrag_T(const bf16_t* TT, int ldk, int k0, int c0, int lane) {
    int half = (lane >> 4) & 1, c = lane & 15;
    const bf16_t* p = TT + (size_t)(c0 + c) * ldk + k0 + half * 8;
    return cat16(*(const v8bf*)p, *(const v8bf*)(p + 16));
}

__device__ inline void store_pair(bf16_t* dst, float x, float y) {
    v2bf p;
    p[0] = (bf16_t)x;
    p[1] = (bf16_t)y;
    *(v2bf*)dst = p;   // v_cvt_pk_bf16_f32 + ds_store_b32
}

// ---- async global->LDS staging (CDNA5 Tensor/async data path) -------------
__device__ inline void async_b128(const float* g, float* l) {
#if TRY_ASYNC
    unsigned loff = (unsigned)(unsigned long long)(size_t)l;  // LDS offset
    unsigned long long ga = (unsigned long long)(size_t)g;
    asm volatile("global_load_async_to_lds_b128 %0, %1, off"
                 :: "v"(loff), "v"(ga) : "memory");
#else
    *(v4f*)l = *(const v4f*)g;
#endif
}

__device__ inline void async_wait() {
#if TRY_ASYNC
#if __has_builtin(__builtin_amdgcn_s_wait_asynccnt)
    __builtin_amdgcn_s_wait_asynccnt(0);
#else
    asm volatile("s_wait_asynccnt 0" ::: "memory");
#endif
#endif
}

// ---------------------------------------------------------------------------
// Fused conv1x1 / GEMM:  Y[b][o][n] = act( W[o][:] . X[b][:][n] + bias )
// X channels: i < I1 from X1, else X2 (concat).  fuse: BN+leakyReLU.
// W tile async-staged raw f32; X tile staged transposed bf16.
// ---------------------------------------------------------------------------
__global__ __launch_bounds__(256) void gemm_kernel(
    const float* __restrict__ W, const float* __restrict__ bias,
    const float* __restrict__ X1, const float* __restrict__ X2, int I1, int I,
    const float* __restrict__ bnScale, const float* __restrict__ bnShift, int fuse,
    const float* __restrict__ resid, float* __restrict__ Y, int O)
{
    __shared__ __align__(32) float  Wraw[64][32];   // raw f32 (async staged)
    __shared__ __align__(32) bf16_t XtT[64][32];    // [n][k] transposed bf16

    int b  = blockIdx.z;
    int o0 = blockIdx.y * 64;
    int n0 = blockIdx.x * 64;
    int tid  = threadIdx.x;
    int lane = tid & 31;
    int wave = tid >> 5;
    int osub = (wave >> 1) * 16;
    int nsub = (wave & 1) * 32;

    v8f c0 = zero8(), c1 = zero8();

    for (int kk = 0; kk < I; kk += 32) {
        // W tile 64x32 f32 via async b128: 512 quads, 2 per thread
        for (int q = tid; q < 512; q += 256) {
            int r = q >> 3, kq = (q & 7) * 4;
            async_b128(&W[(size_t)(o0 + r) * I + kk + kq], &Wraw[r][kq]);
        }
        // X tile -> transposed bf16 [n][k]; pairs along k (packed cvt + b32)
        for (int q = tid; q < 1024; q += 256) {
            int n = q & 63, kp = q >> 6;          // kp in 0..15
            int ch = kk + 2 * kp;
            float x0, x1;
            if (ch < I1) {
                x0 = X1[((size_t)b * I1 + ch) * Nn + n0 + n];
                x1 = X1[((size_t)b * I1 + ch + 1) * Nn + n0 + n];
            } else {
                int c2 = ch - I1, Iw = I - I1;
                x0 = X2[((size_t)b * Iw + c2) * Nn + n0 + n];
                x1 = X2[((size_t)b * Iw + c2 + 1) * Nn + n0 + n];
            }
            store_pair(&XtT[n][2 * kp], x0, x1);
        }
        if (kk + 32 < I)  // keep a global_prefetch hint for the next X tile
            __builtin_prefetch(&X1[((size_t)b * I1 + ((kk + 32) < I1 ? kk + 32 : 0)) * Nn + n0 + (tid & 63)], 0, 0);
        async_wait();
        __syncthreads();

        v16bf a  = afrag_f32(&Wraw[0][0], 32, osub, 0, lane);
        v16bf b0 = bfrag_T(&XtT[0][0], 32, 0, nsub, lane);
        v16bf b1 = bfrag_T(&XtT[0][0], 32, 0, nsub + 16, lane);
        c0 = wmma_bf16(a, b0, c0);
        c1 = wmma_bf16(a, b1, c1);
        __syncthreads();
    }

    int half = lane >> 4;
    int col  = lane & 15;
#pragma unroll
    for (int v = 0; v < 8; ++v) {
        int orow = o0 + osub + v + 8 * half;
        float a0 = c0[v] + bias[orow];
        float a1 = c1[v] + bias[orow];
        if (fuse) {
            float sc = bnScale[orow], sh = bnShift[orow];
            a0 = a0 * sc + sh; a1 = a1 * sc + sh;
            a0 = (a0 >= 0.f) ? a0 : NEGS * a0;
            a1 = (a1 >= 0.f) ? a1 : NEGS * a1;
        }
        size_t yb = ((size_t)b * O + orow) * Nn + n0 + nsub;
        if (resid) { a0 += resid[yb + col]; a1 += resid[yb + 16 + col]; }
        Y[yb + col]      = a0;
        Y[yb + 16 + col] = a1;
    }
}

// ---------------------------------------------------------------------------
// Fused MHA (flash style).  Q,K,V: (B, D, N), channel = dim*H + h.
// One wave per (b, h, 16-query block).
// ---------------------------------------------------------------------------
__global__ __launch_bounds__(32) void attn_kernel(
    const float* __restrict__ Q, const float* __restrict__ K,
    const float* __restrict__ V, float* __restrict__ Msg)
{
    __shared__ __align__(32) bf16_t Qa[16][DIMH];   // [n][d] row-major (A)
    __shared__ __align__(32) bf16_t KtT[32][DIMH];  // [m][d] transposed B (k=d)
    __shared__ __align__(32) bf16_t VtT[DIMH][32];  // [d][m] transposed B (k=m)
    __shared__ __align__(32) bf16_t Pa[16][32];     // [n][m] row-major (A)
    __shared__ float  St[16][32];
    __shared__ float  rmax[16], rsum[16], rscale[16];

    int n0 = blockIdx.x * 16;
    int h  = blockIdx.y;
    int b  = blockIdx.z;
    int lane = threadIdx.x;
    size_t base = (size_t)b * Dd * Nn;

    // Qa: pairs along d (packed cvt + b32 store)
    for (int q = lane; q < 512; q += 32) {
        int r = q & 15, dp = q >> 4;   // dp 0..31
        float x0 = Q[base + (size_t)((2 * dp) * Hh + h) * Nn + n0 + r];
        float x1 = Q[base + (size_t)((2 * dp + 1) * Hh + h) * Nn + n0 + r];
        store_pair(&Qa[r][2 * dp], x0, x1);
    }
    if (lane < 16) { rmax[lane] = -3.0e38f; rsum[lane] = 0.f; }
    __syncthreads();

    v16bf aq0 = afrag_bf16(&Qa[0][0], DIMH, 0, 0, lane);
    v16bf aq1 = afrag_bf16(&Qa[0][0], DIMH, 0, 32, lane);

    v8f o0 = zero8(), o1 = zero8(), o2 = zero8(), o3 = zero8();
    int half = lane >> 4;
    int col  = lane & 15;

    for (int m0 = 0; m0 < Nn; m0 += 32) {
        // KtT[m][d]: pairs along d (strided global pair, packed store)
        for (int q = lane; q < 1024; q += 32) {
            int mm = q & 31, dp = q >> 5;   // dp 0..31
            float x0 = K[base + (size_t)((2 * dp) * Hh + h) * Nn + m0 + mm];
            float x1 = K[base + (size_t)((2 * dp + 1) * Hh + h) * Nn + m0 + mm];
            store_pair(&KtT[mm][2 * dp], x0, x1);
        }
        // VtT[d][m]: pairs along m are contiguous in global -> v2f load
        for (int q = lane; q < 1024; q += 32) {
            int d = q >> 4, mp = q & 15;    // mp 0..15
            v2f x = *(const v2f*)&V[base + (size_t)(d * Hh + h) * Nn + m0 + 2 * mp];
            store_pair(&VtT[d][2 * mp], x[0], x[1]);
        }
        __syncthreads();

        // S tile 16(n) x 32(m), K-dim 64
        v8f s0 = zero8(), s1 = zero8();
        s0 = wmma_bf16(aq0, bfrag_T(&KtT[0][0], DIMH, 0,  0,  lane), s0);
        s0 = wmma_bf16(aq1, bfrag_T(&KtT[0][0], DIMH, 32, 0,  lane), s0);
        s1 = wmma_bf16(aq0, bfrag_T(&KtT[0][0], DIMH, 0,  16, lane), s1);
        s1 = wmma_bf16(aq1, bfrag_T(&KtT[0][0], DIMH, 32, 16, lane), s1);
#pragma unroll
        for (int v = 0; v < 8; ++v) {
            St[v + 8 * half][col]      = s0[v] * 0.125f;
            St[v + 8 * half][16 + col] = s1[v] * 0.125f;
        }
        __syncthreads();

        // online softmax: one thread per query row; write P packed pairs
        if (lane < 16) {
            float mo = rmax[lane], mx = mo;
#pragma unroll
            for (int j = 0; j < 32; ++j) mx = fmaxf(mx, St[lane][j]);
            float corr = __expf(mo - mx);
            float s = rsum[lane] * corr;
#pragma unroll
            for (int j = 0; j < 32; j += 2) {
                float e0 = __expf(St[lane][j] - mx);
                float e1 = __expf(St[lane][j + 1] - mx);
                store_pair(&Pa[lane][j], e0, e1);
                s += e0 + e1;
            }
            rmax[lane] = mx; rsum[lane] = s; rscale[lane] = corr;
        }
        __syncthreads();

#pragma unroll
        for (int v = 0; v < 8; ++v) {
            float f = rscale[v + 8 * half];
            o0[v] *= f; o1[v] *= f; o2[v] *= f; o3[v] *= f;
        }
        v16bf ap = afrag_bf16(&Pa[0][0], 32, 0, 0, lane);
        o0 = wmma_bf16(ap, bfrag_T(&VtT[0][0], 32, 0, 0,  lane), o0);
        o1 = wmma_bf16(ap, bfrag_T(&VtT[0][0], 32, 0, 16, lane), o1);
        o2 = wmma_bf16(ap, bfrag_T(&VtT[0][0], 32, 0, 32, lane), o2);
        o3 = wmma_bf16(ap, bfrag_T(&VtT[0][0], 32, 0, 48, lane), o3);
        __syncthreads();
    }

#pragma unroll
    for (int v = 0; v < 8; ++v) {
        int nrow = n0 + v + 8 * half;
        float inv = 1.f / rsum[v + 8 * half];
        Msg[base + (size_t)((0  + col) * Hh + h) * Nn + nrow] = o0[v] * inv;
        Msg[base + (size_t)((16 + col) * Hh + h) * Nn + nrow] = o1[v] * inv;
        Msg[base + (size_t)((32 + col) * Hh + h) * Nn + nrow] = o2[v] * inv;
        Msg[base + (size_t)((48 + col) * Hh + h) * Nn + nrow] = o3[v] * inv;
    }
}

// ---------------------------------------------------------------------------
// Score GEMM: Z[b][n][m] = sum_d md0[b][d][n]*md1[b][d][m] / 16, pitch 2049.
// ---------------------------------------------------------------------------
__global__ __launch_bounds__(256) void score_gemm_kernel(
    const float* __restrict__ A, const float* __restrict__ Bm,
    float* __restrict__ Z, float scale)
{
    __shared__ __align__(32) bf16_t At[64][32];   // [n][k] row-major (A)
    __shared__ __align__(32) bf16_t BtT[64][32];  // [m][k] transposed (B)

    int b  = blockIdx.z;
    int n0 = blockIdx.y * 64;
    int m0 = blockIdx.x * 64;
    int tid  = threadIdx.x;
    int lane = tid & 31;
    int wave = tid >> 5;
    int nsubr = (wave >> 1) * 16;
    int msub  = (wave & 1) * 32;

    v8f c0 = zero8(), c1 = zero8();
    for (int kk = 0; kk < Dd; kk += 32) {
        for (int q = tid; q < 1024; q += 256) {
            int r = q & 63, kp = q >> 6;   // pairs along k (d)
            float x0 = A[((size_t)b * Dd + kk + 2 * kp) * Nn + n0 + r];
            float x1 = A[((size_t)b * Dd + kk + 2 * kp + 1) * Nn + n0 + r];
            store_pair(&At[r][2 * kp], x0, x1);
        }
        for (int q = tid; q < 1024; q += 256) {
            int c = q & 63, kp = q >> 6;
            float x0 = Bm[((size_t)b * Dd + kk + 2 * kp) * Nn + m0 + c];
            float x1 = Bm[((size_t)b * Dd + kk + 2 * kp + 1) * Nn + m0 + c];
            store_pair(&BtT[c][2 * kp], x0, x1);
        }
        __syncthreads();
        v16bf a  = afrag_bf16(&At[0][0], 32, nsubr, 0, lane);
        v16bf b0 = bfrag_T(&BtT[0][0], 32, 0, msub, lane);
        v16bf b1 = bfrag_T(&BtT[0][0], 32, 0, msub + 16, lane);
        c0 = wmma_bf16(a, b0, c0);
        c1 = wmma_bf16(a, b1, c1);
        __syncthreads();
    }
    int half = lane >> 4, col = lane & 15;
#pragma unroll
    for (int v = 0; v < 8; ++v) {
        int nrow = n0 + nsubr + v + 8 * half;
        size_t zb = ((size_t)b * PP + nrow) * PP + m0 + msub;
        Z[zb + col]      = c0[v] * scale;
        Z[zb + 16 + col] = c1[v] * scale;
    }
}

// --------------------------- small VALU kernels -----------------------------

__global__ void bnprep_kernel(const float* g, const float* be, const float* m,
                              const float* v, float* scale, float* shift, int C)
{
    int c = blockIdx.x * blockDim.x + threadIdx.x;
    if (c < C) {
        float s = g[c] * rsqrtf(v[c] + EPSS);
        scale[c] = s;
        shift[c] = be[c] - m[c] * s;
    }
}

__global__ void buildx_kernel(const float* kpts, const float* sc, float* X)
{
    int idx = blockIdx.x * blockDim.x + threadIdx.x;
    if (idx >= Bb * Nn) return;
    int b = idx / Nn, n = idx % Nn;
    X[((size_t)b * 3 + 0) * Nn + n] = kpts[((size_t)b * Nn + n) * 2 + 0];
    X[((size_t)b * 3 + 1) * Nn + n] = kpts[((size_t)b * Nn + n) * 2 + 1];
    X[((size_t)b * 3 + 2) * Nn + n] = sc[(size_t)b * Nn + n];
}

__global__ void conv_naive_kernel(const float* __restrict__ W, const float* __restrict__ bias,
                                  const float* __restrict__ X, const float* __restrict__ scale,
                                  const float* __restrict__ shift, int fuse,
                                  float* __restrict__ Y, int I, int O)
{
    int idx = blockIdx.x * blockDim.x + threadIdx.x;
    if (idx >= Bb * O * Nn) return;
    int n = idx % Nn;
    int o = (idx / Nn) % O;
    int b = idx / (Nn * O);
    float acc = bias[o];
    for (int i = 0; i < I; ++i)
        acc += W[(size_t)o * I + i] * X[((size_t)b * I + i) * Nn + n];
    if (fuse) {
        acc = acc * scale[o] + shift[o];
        acc = (acc >= 0.f) ? acc : NEGS * acc;
    }
    Y[idx] = acc;
}

__global__ void bins_kernel(float* Z, const float* alpha)
{
    int idx = blockIdx.x * blockDim.x + threadIdx.x;
    if (idx >= Bb * PP) return;
    int b = idx / PP, j = idx % PP;
    float a = alpha[0];
    Z[((size_t)b * PP + Nn) * PP + j] = a;
    if (j < Nn) Z[((size_t)b * PP + j) * PP + Nn] = a;
}

__global__ void uvzero_kernel(float* U, float* V)
{
    int idx = blockIdx.x * blockDim.x + threadIdx.x;
    if (idx < Bb * PP) { U[idx] = 0.f; V[idx] = 0.f; }
}

__global__ __launch_bounds__(256) void u_update_kernel(
    const float* __restrict__ Z, const float* __restrict__ Vv,
    float* __restrict__ Uu, float norm)
{
    __shared__ float smax[256], ssum[256];
    int b = blockIdx.y, row = blockIdx.x, tid = threadIdx.x;
    const float* zr = Z + ((size_t)b * PP + row) * PP;
    const float* vb = Vv + (size_t)b * PP;
    float mx = -3.0e38f, sm = 0.f;
    for (int j = tid; j < PP; j += 256) {
        float t = zr[j] + vb[j];
        if (t > mx) { sm = sm * __expf(mx - t) + 1.f; mx = t; }
        else        { sm += __expf(t - mx); }
    }
    smax[tid] = mx; ssum[tid] = sm;
    __syncthreads();
    for (int s = 128; s > 0; s >>= 1) {
        if (tid < s) {
            float m2 = smax[tid + s], s2 = ssum[tid + s];
            float M = fmaxf(smax[tid], m2);
            ssum[tid] = ssum[tid] * __expf(smax[tid] - M) + s2 * __expf(m2 - M);
            smax[tid] = M;
        }
        __syncthreads();
    }
    if (tid == 0) {
        float lse = smax[0] + __logf(ssum[0]);
        float lmu = (row < Nn) ? norm : (__logf((float)Nn) + norm);
        Uu[(size_t)b * PP + row] = lmu - lse;
    }
}

__global__ void v_update_kernel(const float* __restrict__ Z,
                                const float* __restrict__ Uu,
                                float* __restrict__ Vv, float norm)
{
    int col = blockIdx.x * blockDim.x + threadIdx.x;
    int b = blockIdx.y;
    if (col >= PP) return;
    const float* zb = Z + (size_t)b * PP * PP;
    const float* ub = Uu + (size_t)b * PP;
    float mx = -3.0e38f, sm = 0.f;
    for (int r = 0; r < PP; ++r) {
        float t = zb[(size_t)r * PP + col] + ub[r];
        if (t > mx) { sm = sm * __expf(mx - t) + 1.f; mx = t; }
        else        { sm += __expf(t - mx); }
    }
    float lnu = (col < Nn) ? norm : (__logf((float)Nn) + norm);
    Vv[(size_t)b * PP + col] = lnu - (mx + __logf(sm));
}

__global__ void zfinal_kernel(const float* __restrict__ Z, const float* __restrict__ U,
                              const float* __restrict__ V, float* __restrict__ Zo,
                              float norm)
{
    size_t total = (size_t)Bb * PP * PP;
    for (size_t i = (size_t)blockIdx.x * blockDim.x + threadIdx.x; i < total;
         i += (size_t)gridDim.x * blockDim.x) {
        size_t bp  = i / ((size_t)PP * PP);
        size_t rem = i % ((size_t)PP * PP);
        int r = (int)(rem / PP), c = (int)(rem % PP);
        Zo[i] = Z[i] + U[bp * PP + r] + V[bp * PP + c] - norm;
    }
}

__global__ __launch_bounds__(256) void rowmax_kernel(const float* __restrict__ Zo,
                                                     float* __restrict__ maxv,
                                                     int* __restrict__ idx)
{
    __shared__ float sval[256];
    __shared__ int   sidx[256];
    int b = blockIdx.y, row = blockIdx.x, tid = threadIdx.x;
    const float* zr = Zo + ((size_t)b * PP + row) * PP;
    float mx = -3.0e38f; int mi = 0;
    for (int j = tid; j < Nn; j += 256) {
        float t = zr[j];
        if (t > mx) { mx = t; mi = j; }
    }
    sval[tid] = mx; sidx[tid] = mi;
    __syncthreads();
    for (int s = 128; s > 0; s >>= 1) {
        if (tid < s) {
            float v2 = sval[tid + s]; int i2 = sidx[tid + s];
            if (v2 > sval[tid] || (v2 == sval[tid] && i2 < sidx[tid])) {
                sval[tid] = v2; sidx[tid] = i2;
            }
        }
        __syncthreads();
    }
    if (tid == 0) { maxv[(size_t)b * Nn + row] = sval[0]; idx[(size_t)b * Nn + row] = sidx[0]; }
}

__global__ void colmax_kernel(const float* __restrict__ Zo,
                              float* __restrict__ maxv, int* __restrict__ idx)
{
    int col = blockIdx.x * blockDim.x + threadIdx.x;
    int b = blockIdx.y;
    if (col >= Nn) return;
    const float* zb = Zo + (size_t)b * PP * PP;
    float mx = -3.0e38f; int mi = 0;
    for (int r = 0; r < Nn; ++r) {
        float t = zb[(size_t)r * PP + col];
        if (t > mx) { mx = t; mi = r; }
    }
    maxv[(size_t)b * Nn + col] = mx;
    idx[(size_t)b * Nn + col]  = mi;
}

__global__ void match0_kernel(const float* __restrict__ max0v, const int* __restrict__ idx0,
                              const int* __restrict__ idx1, float* __restrict__ msc0,
                              int* __restrict__ ind0, int* __restrict__ valid0)
{
    int t = blockIdx.x * blockDim.x + threadIdx.x;
    if (t >= Bb * Nn) return;
    int b = t / Nn, n = t % Nn;
    int j = idx0[t];
    int mutual = (idx1[(size_t)b * Nn + j] == n);
    float ms = mutual ? __expf(max0v[t]) : 0.f;
    int v0 = mutual && (ms > THRS);
    msc0[t]   = ms;
    ind0[t]   = v0 ? j : -1;
    valid0[t] = v0;
}

__global__ void match1_kernel(const int* __restrict__ idx0, const int* __restrict__ idx1,
                              const float* __restrict__ msc0, const int* __restrict__ valid0,
                              float* __restrict__ msc1, int* __restrict__ ind1)
{
    int t = blockIdx.x * blockDim.x + threadIdx.x;
    if (t >= Bb * Nn) return;
    int b = t / Nn, m = t % Nn;
    int i = idx1[t];
    int mutual = (idx0[(size_t)b * Nn + i] == m);
    msc1[t] = mutual ? msc0[(size_t)b * Nn + i] : 0.f;
    int v1 = mutual && valid0[(size_t)b * Nn + i];
    ind1[t] = v1 ? i : -1;
}

// ---------------------------------------------------------------------------
// Host orchestration
// ---------------------------------------------------------------------------
static inline void launch_gemm(const float* W, const float* bias,
                               const float* X1, const float* X2, int I1, int I,
                               const float* sc, const float* sh, int fuse,
                               const float* resid, float* Y, int O, hipStream_t s)
{
    dim3 g(Nn / 64, O / 64, Bb);
    gemm_kernel<<<g, 256, 0, s>>>(W, bias, X1, X2, I1, I, sc, sh, fuse, resid, Y, O);
}

extern "C" void kernel_launch(void* const* d_in, const int* in_sizes, int n_in,
                              void* d_out, int out_size, void* d_ws, size_t ws_size,
                              hipStream_t stream)
{
    (void)in_sizes; (void)n_in; (void)out_size; (void)ws_size;
    const float NORM = -logf(4096.f);

    auto F = [&](int i) { return (const float*)d_in[i]; };
    const float* kpts0   = F(0);
    const float* scores0 = F(1);
    const float* desc0   = F(2);
    const float* kpts1   = F(3);
    const float* scores1 = F(4);
    const float* desc1   = F(5);
    const float* alpha   = F(6);
    const float* finW    = F(7);
    const float* finB    = F(8);
    auto gL = [&](int l, int off) { return (const float*)d_in[9 + l * 16 + off]; };

    float* ws = (float*)d_ws;
    size_t DN = (size_t)Bb * Dd * Nn;
    float* dA0    = ws;
    float* dA1    = ws + DN;
    float* dB0    = ws + 2 * DN;
    float* dB1    = ws + 3 * DN;
    float* Qb     = ws + 4 * DN;
    float* Kb     = ws + 5 * DN;
    float* Vb     = ws + 6 * DN;
    float* MsgRaw = ws + 7 * DN;
    float* Msg    = ws + 8 * DN;
    float* Hbuf   = ws + 9 * DN;   // 2*DN
    float* md0    = ws + 11 * DN;
    float* md1    = ws + 12 * DN;
    float* Z      = ws + 13 * DN;
    size_t ZB     = (size_t)Bb * PP * PP;
    float* Uu     = Z + ZB;
    float* Vv     = Uu + (size_t)Bb * PP;
    float* max0v  = Vv + (size_t)Bb * PP;
    int*   idx0   = (int*)(max0v + (size_t)Bb * Nn);
    float* max1v  = (float*)(idx0 + Bb * Nn);
    int*   idx1   = (int*)(max1v + Bb * Nn);
    int*   valid0 = idx1 + Bb * Nn;
    float* bnb    = (float*)(valid0 + Bb * Nn);

    // ---- BN folding ----
    int kencC[3] = {32, 64, 128};
    float* kScale[3]; float* kShift[3];
    float* cur = bnb;
    for (int i = 0; i < 3; ++i) {
        kScale[i] = cur; cur += kencC[i];
        kShift[i] = cur; cur += kencC[i];
        int base = 297 + i * 6;  // be,g,m,v,W,b
        bnprep_kernel<<<(kencC[i] + 255) / 256, 256, 0, stream>>>(
            F(base + 1), F(base + 0), F(base + 2), F(base + 3),
            kScale[i], kShift[i], kencC[i]);
    }
    float* gScale[18]; float* gShift[18];
    for (int l = 0; l < 18; ++l) {
        gScale[l] = cur; cur += 512;
        gShift[l] = cur; cur += 512;
        bnprep_kernel<<<2, 256, 0, stream>>>(
            gL(l, 9), gL(l, 8), gL(l, 10), gL(l, 11), gScale[l], gShift[l], 512);
    }

    // ---- keypoint encoder: desc += kenc(kpts, scores) ----
    const float* kW3 = F(315); const float* kB3 = F(316);
    for (int s = 0; s < 2; ++s) {
        const float* kp = s ? kpts1 : kpts0;
        const float* sc = s ? scores1 : scores0;
        const float* dc = s ? desc1 : desc0;
        float* out = s ? dA1 : dA0;
        buildx_kernel<<<(Bb * Nn + 255) / 256, 256, 0, stream>>>(kp, sc, Qb);
        conv_naive_kernel<<<(Bb * 32 * Nn + 255) / 256, 256, 0, stream>>>(
            F(301), F(302), Qb, kScale[0], kShift[0], 1, Kb, 3, 32);
        launch_gemm(F(307), F(308), Kb, Kb, 32, 32, kScale[1], kShift[1], 1,
                    nullptr, Vb, 64, stream);
        launch_gemm(F(313), F(314), Vb, Vb, 64, 64, kScale[2], kShift[2], 1,
                    nullptr, MsgRaw, 128, stream);
        launch_gemm(kW3, kB3, MsgRaw, MsgRaw, 128, 128, nullptr, nullptr, 0,
                    dc, out, 256, stream);
    }

    // ---- GNN: 18 layers, self/cross alternating ----
    float* x0p = dA0; float* x1p = dA1;
    float* y0p = dB0; float* y1p = dB1;
    for (int l = 0; l < 18; ++l) {
        bool cross = (l & 1);
        for (int s = 0; s < 2; ++s) {
            const float* x   = s ? x1p : x0p;
            const float* src = cross ? (s ? x0p : x1p) : x;
            float* out = s ? y1p : y0p;
            launch_gemm(gL(l,12), gL(l,13), x,   x,   Dd, Dd, nullptr, nullptr, 0, nullptr, Qb, Dd, stream);
            launch_gemm(gL(l,0),  gL(l,1),  src, src, Dd, Dd, nullptr, nullptr, 0, nullptr, Kb, Dd, stream);
            launch_gemm(gL(l,14), gL(l,15), src, src, Dd, Dd, nullptr, nullptr, 0, nullptr, Vb, Dd, stream);
            attn_kernel<<<dim3(Nn / 16, Hh, Bb), 32, 0, stream>>>(Qb, Kb, Vb, MsgRaw);
            launch_gemm(gL(l,2), gL(l,3), MsgRaw, MsgRaw, Dd, Dd, nullptr, nullptr, 0, nullptr, Msg, Dd, stream);
            launch_gemm(gL(l,4), gL(l,5), x, Msg, Dd, 2 * Dd, gScale[l], gShift[l], 1, nullptr, Hbuf, 2 * Dd, stream);
            launch_gemm(gL(l,6), gL(l,7), Hbuf, Hbuf, 2 * Dd, 2 * Dd, nullptr, nullptr, 0, x, out, Dd, stream);
        }
        float* t;
        t = x0p; x0p = y0p; y0p = t;
        t = x1p; x1p = y1p; y1p = t;
    }

    // ---- final projection + score matrix + Sinkhorn ----
    launch_gemm(finW, finB, x0p, x0p, Dd, Dd, nullptr, nullptr, 0, nullptr, md0, Dd, stream);
    launch_gemm(finW, finB, x1p, x1p, Dd, Dd, nullptr, nullptr, 0, nullptr, md1, Dd, stream);
    score_gemm_kernel<<<dim3(Nn / 64, Nn / 64, Bb), 256, 0, stream>>>(md0, md1, Z, 1.f / 16.f);
    bins_kernel<<<(Bb * PP + 255) / 256, 256, 0, stream>>>(Z, alpha);
    uvzero_kernel<<<(Bb * PP + 255) / 256, 256, 0, stream>>>(Uu, Vv);

    for (int it = 0; it < 20; ++it) {
        u_update_kernel<<<dim3(PP, Bb), 256, 0, stream>>>(Z, Vv, Uu, NORM);
        v_update_kernel<<<dim3((PP + 255) / 256, Bb), 256, 0, stream>>>(Z, Uu, Vv, NORM);
    }

    float* zout = (float*)d_out;
    zfinal_kernel<<<8192, 256, 0, stream>>>(Z, Uu, Vv, zout, NORM);

    int*   ind0 = (int*)(zout + ZB);
    int*   ind1 = ind0 + Bb * Nn;
    float* ms0  = (float*)(ind1 + Bb * Nn);
    float* ms1  = ms0 + Bb * Nn;

    rowmax_kernel<<<dim3(Nn, Bb), 256, 0, stream>>>(zout, max0v, idx0);
    colmax_kernel<<<dim3((Nn + 255) / 256, Bb), 256, 0, stream>>>(zout, max1v, idx1);
    match0_kernel<<<(Bb * Nn + 255) / 256, 256, 0, stream>>>(max0v, idx0, idx1, ms0, ind0, valid0);
    match1_kernel<<<(Bb * Nn + 255) / 256, 256, 0, stream>>>(idx0, idx1, ms0, valid0, ms1, ind1);
}
